// MHSAWithPositionEmbedding_61761629716971
// MI455X (gfx1250) — matrix operand order repaired and checked by reading, not compiled, not run on hardware
//
#include <hip/hip_runtime.h>

// ---------------------------------------------------------------------------
// MHSA + 2D relative position bias for MI455X (gfx1250, wave32, WMMA bf16).
// ~35 GFLOP, flash-attention fusion keeps HBM traffic ~60MB -> matrix-core
// bound. All matmuls run on v_wmma_f32_16x16x32_bf16 (f32 accum). fp32->bf16
// conversion is done ONCE in pre-pass kernels, so every GEMM tile stage is a
// pure b128 copy -> issued as GLOBAL_LOAD_ASYNC_TO_LDS_B128 (ASYNCcnt), no
// VGPR staging, single s_wait_asynccnt before the barrier.
// ---------------------------------------------------------------------------

#define HEADS 4
#define KD    128
#define LSEQ  1024

typedef __attribute__((ext_vector_type(16))) __bf16 v16bf;
typedef __attribute__((ext_vector_type(8)))  float  v8f;

union Frag128 { uint4 q[2]; unsigned int u[8]; v16bf v; };

__device__ __forceinline__ unsigned int pack2_bf16(float a, float b) {
  unsigned int ua = __float_as_uint(a);
  unsigned int ub = __float_as_uint(b);
  ua = (ua + 0x7FFFu + ((ua >> 16) & 1u)) >> 16;   // round-to-nearest-even
  ub = (ub + 0x7FFFu + ((ub >> 16) & 1u)) >> 16;
  return (ua & 0xFFFFu) | (ub << 16);
}
__device__ __forceinline__ unsigned short f2bf(float a) {
  unsigned int ua = __float_as_uint(a);
  return (unsigned short)((ua + 0x7FFFu + ((ua >> 16) & 1u)) >> 16);
}

__device__ __forceinline__ v8f wmma_bf16(v16bf a, v16bf b, v8f c) {
  return __builtin_amdgcn_wmma_f32_16x16x32_bf16(false, a, false, b, (short)0, c,
                                                 false, false);
}

// Async global -> LDS 128-bit copy (gfx1250 GLOBAL_LOAD_ASYNC_TO_LDS_B128).
// LDS destination offset = low 32 bits of the generic pointer (ISA 10.2:
// LDS aperture address truncates to the in-wave LDS byte offset).
__device__ __forceinline__ void async_g2l_b128(unsigned int* lds_dst,
                                               const unsigned int* gsrc) {
  unsigned int lds = (unsigned int)(size_t)lds_dst;
  asm volatile("global_load_async_to_lds_b128 %0, %1, off"
               :: "v"(lds), "v"(gsrc) : "memory");
}
__device__ __forceinline__ void wait_async0() {
  asm volatile("s_wait_asynccnt 0x0" ::: "memory");
}

// A-operand (16x32 bf16): lane row M = lane&15; pair indices
// {pairbase + half*4 + 0..3 , pairbase + 8 + half*4 + 0..3} -> two b128 loads.
__device__ __forceinline__ v16bf load_a_frag(const unsigned int* base, int row,
                                             int pitchU, int pairbase, int lane) {
  const unsigned int* p = base + row * pitchU + pairbase + ((lane >> 4) & 1) * 4;
  Frag128 f;
  f.q[0] = *(const uint4*)(p);
  f.q[1] = *(const uint4*)(p + 8);
  return f.v;
}

// B-operand (32x16 bf16) from K-transposed tile [n][K-pair]: lanes 0-15 take
// K-pairs {0..3, 8..11}, lanes 16-31 take {4..7, 12..15} -> two b128 loads.
__device__ __forceinline__ v16bf load_b_frag(const unsigned int* base, int nbase,
                                             int pitchU, int pairbase, int lane) {
  const unsigned int* p =
      base + (nbase + (lane & 15)) * pitchU + pairbase + ((lane >> 4) & 1) * 4;
  Frag128 f;
  f.q[0] = *(const uint4*)(p);
  f.q[1] = *(const uint4*)(p + 8);
  return f.v;
}

// A-fragment straight from a packed-bf16 row (row pitch supplied by caller).
__device__ __forceinline__ v16bf load_a_frag_row(const unsigned int* row,
                                                 int kk, int lane) {
  const unsigned int* p = row + kk * 16 + ((lane >> 4) & 1) * 4;
  Frag128 f;
  f.q[0] = *(const uint4*)(p);
  f.q[1] = *(const uint4*)(p + 8);
  return f.v;
}

// ---------------------------------------------------------------------------
// Pre-pass 1: pack fp32 rows into bf16 pairs (x -> xp). One float4 per thread.
// ---------------------------------------------------------------------------
__global__ __launch_bounds__(256)
void pack_rows_kernel(const float* __restrict__ src, uint2* __restrict__ dst) {
  size_t idx = (size_t)blockIdx.x * 256 + threadIdx.x;
  float4 f = ((const float4*)src)[idx];
  dst[idx] = make_uint2(pack2_bf16(f.x, f.y), pack2_bf16(f.z, f.w));
}

// ---------------------------------------------------------------------------
// Pre-pass 2: W[512x512] fp32 -> Wt[n][k-pair] packed bf16 (B-operand ready).
// 64x64 LDS tile transpose; coalesced loads and stores.
// ---------------------------------------------------------------------------
__global__ __launch_bounds__(256)
void pack_transpose_kernel(const float* __restrict__ W,
                           unsigned int* __restrict__ Wt) {
  __shared__ float tile[64 * 65];
  const int tid = threadIdx.x;
  const int k0 = blockIdx.x * 64;
  const int n0 = blockIdx.y * 64;
#pragma unroll
  for (int i = 0; i < 16; ++i) {
    int q = tid + 256 * i;                   // 64 k x 64 n
    int kk = q >> 6, nn = q & 63;
    tile[kk * 65 + nn] = W[(size_t)(k0 + kk) * 512 + n0 + nn];
  }
  __syncthreads();
#pragma unroll
  for (int i = 0; i < 8; ++i) {
    int q = tid + 256 * i;                   // 64 n x 32 pairs
    int n = q >> 5, pp = q & 31;
    unsigned int val = pack2_bf16(tile[(2 * pp) * 65 + n],
                                  tile[(2 * pp + 1) * 65 + n]);
    Wt[(size_t)(n0 + n) * 256 + (k0 >> 1) + pp] = val;
  }
}

// ---------------------------------------------------------------------------
// GEMM: C[128x128 tile] = A[M x 512] * B[512 x 512], bf16 WMMA, f32 accum.
// Tiles are staged with async global->LDS b128 copies.
// MODE 0: q/k epilogue ([b][h][l][d] bf16, scaled)
// MODE 1: v epilogue, pre-transposed ([bh][d][l-pair] packed bf16)
// MODE 2: fp32 row-major (final output)
// ---------------------------------------------------------------------------
template <int MODE>
__global__ __launch_bounds__(256)
void gemm_wmma_kernel(const unsigned int* __restrict__ Ap,  // [M][256 pairs]
                      const unsigned int* __restrict__ Bt,  // [512][256 pairs]
                      float scale,
                      unsigned short* __restrict__ out_qkv,
                      unsigned int* __restrict__ out_vt,
                      float* __restrict__ out_f32) {
  __shared__ __align__(16) unsigned int Alds[128 * 20];  // [row][16 pairs]+pad
  __shared__ __align__(16) unsigned int Blds[128 * 20];  // [n][16 pairs]+pad

  const int tid  = threadIdx.x;
  const int lane = tid & 31;
  const int wave = tid >> 5;
  const int wm   = wave >> 1;               // 4 waves over M
  const int wn   = wave & 1;                // 2 waves over N
  const int m0   = blockIdx.x * 128;
  const int n0   = blockIdx.y * 128;

  const v8f zero = {0.f, 0.f, 0.f, 0.f, 0.f, 0.f, 0.f, 0.f};
  v8f acc[2][4];
#pragma unroll
  for (int i = 0; i < 2; ++i)
#pragma unroll
    for (int p = 0; p < 4; ++p) acc[i][p] = zero;

  for (int kk = 0; kk < 512; kk += 32) {
    __syncthreads();
#pragma unroll
    for (int i = 0; i < 2; ++i) {            // A tile: 128 rows x 4 uint4
      int q = tid + 256 * i;
      int row = q >> 2, c4 = (q & 3) * 4;
      async_g2l_b128(Alds + row * 20 + c4,
                     Ap + (size_t)(m0 + row) * 256 + (kk >> 1) + c4);
    }
#pragma unroll
    for (int i = 0; i < 2; ++i) {            // B tile: 128 n x 4 uint4
      int q = tid + 256 * i;
      int n = q >> 2, c4 = (q & 3) * 4;
      async_g2l_b128(Blds + n * 20 + c4,
                     Bt + (size_t)(n0 + n) * 256 + (kk >> 1) + c4);
    }
    wait_async0();
    __syncthreads();

    v16bf a0 = load_a_frag(Alds, wm * 32 + (lane & 15), 20, 0, lane);
    v16bf a1 = load_a_frag(Alds, wm * 32 + 16 + (lane & 15), 20, 0, lane);
#pragma unroll
    for (int p = 0; p < 4; ++p) {
      v16bf bf = load_b_frag(Blds, wn * 64 + p * 16, 20, 0, lane);
      acc[0][p] = wmma_bf16(a0, bf, acc[0][p]);
      acc[1][p] = wmma_bf16(a1, bf, acc[1][p]);
    }
  }

  // Epilogue. C/D layout: row m = (lane>>4)*8 + r, col n = lane&15.
  const int half = (lane >> 4) & 1;
#pragma unroll
  for (int i = 0; i < 2; ++i)
#pragma unroll
    for (int p = 0; p < 4; ++p) {
      int n = n0 + wn * 64 + p * 16 + (lane & 15);
      if (MODE == 1) {
        // V: pack adjacent sequence positions (r, r+1) -> [bh][d][l-pair].
#pragma unroll
        for (int r = 0; r < 8; r += 2) {
          int m = m0 + wm * 32 + i * 16 + half * 8 + r;   // even
          int b = m >> 10, l = m & 1023, h = n >> 7, d = n & 127;
          out_vt[((size_t)((b * HEADS + h) * KD + d)) * 512 + (l >> 1)] =
              pack2_bf16(acc[i][p][r], acc[i][p][r + 1]);
        }
      } else {
#pragma unroll
        for (int r = 0; r < 8; ++r) {
          int m = m0 + wm * 32 + i * 16 + half * 8 + r;
          float val = acc[i][p][r] * scale;
          if (MODE == 0) {
            int b = m >> 10, l = m & 1023, h = n >> 7, d = n & 127;
            out_qkv[((size_t)(b * HEADS + h) * LSEQ + l) * KD + d] = f2bf(val);
          } else {
            out_f32[(size_t)m * 512 + n] = val;
          }
        }
      }
    }
}

// ---------------------------------------------------------------------------
// Relative logits on WMMA. rel_to_abs(einsum) == direct gather:
//   rel_w[b,h,y,x,x2] = sum_d q[b,h,y*32+x,d] * pos_emb_w[d, x2-x+31]
// Per block (mode, bh, s): R[32 x 63] = Q[32 x 128] * E[128 x 63(pad 64)],
// 8 waves = 2(M) x 4(N) tile grid, 4 k-steps each; then LDS gather.
// ---------------------------------------------------------------------------
__global__ __launch_bounds__(256)
void rel_logits_kernel(const unsigned int* __restrict__ qU,  // [bh][l][64 pairs]
                       const float* __restrict__ pw,         // [128][63]
                       const float* __restrict__ ph,         // [128][63]
                       float* __restrict__ relw,             // [bh][y][x][x2]
                       float* __restrict__ relh) {           // [bh][y][x][y2]
  __shared__ __align__(16) unsigned int et[64 * 68];  // [col][d-pair] B layout
  __shared__ float Rlds[32 * 65];                     // R tile, padded

  const int tid  = threadIdx.x;
  const int lane = tid & 31;
  const int wave = tid >> 5;
  const int half = (lane >> 4) & 1;
  const int ln   = lane & 15;
  const int mode = blockIdx.y;
  const int bh   = blockIdx.x >> 5;
  const int s    = blockIdx.x & 31;          // fixed coord (y / x)
  const float* emb = mode ? ph : pw;
  float* outb = mode ? relh : relw;

  // Stage E transposed: et[col][dpair]; col 63 is zero padding.
#pragma unroll
  for (int i = 0; i < 16; ++i) {
    int q = tid + 256 * i;                   // 64 dpairs x 64 cols
    int col = q & 63, dp = q >> 6;
    unsigned int val = 0u;
    if (col < 63)
      val = pack2_bf16(emb[(size_t)(2 * dp) * 63 + col],
                       emb[(size_t)(2 * dp + 1) * 63 + col]);
    et[col * 68 + dp] = val;
  }

  // Q fragments: wave tile (wm in M, wn in N).
  const int wm = wave & 1;                   // 2 m-tiles of 16
  const int wn = wave >> 1;                  // 4 n-tiles of 16
  const int i_row = wm * 16 + ln;            // varying image coord
  const int l = mode ? (i_row * 32 + s) : (s * 32 + i_row);
  const unsigned int* qrow = qU + ((size_t)bh * LSEQ + l) * 64;
  v16bf aq[4];
#pragma unroll
  for (int kk = 0; kk < 4; ++kk) aq[kk] = load_a_frag_row(qrow, kk, lane);

  __syncthreads();

  v8f acc = {0.f, 0.f, 0.f, 0.f, 0.f, 0.f, 0.f, 0.f};
#pragma unroll
  for (int kk = 0; kk < 4; ++kk) {
    v16bf bf = load_b_frag(et, wn * 16, 68, kk * 16, lane);
    acc = wmma_bf16(aq[kk], bf, acc);
  }

  // Scatter R tile to LDS, then rel_to_abs gather (branch-free store).
#pragma unroll
  for (int r = 0; r < 8; ++r)
    Rlds[(wm * 16 + half * 8 + r) * 65 + wn * 16 + ln] = acc[r];
  __syncthreads();

#pragma unroll
  for (int rep = 0; rep < 4; ++rep) {
    int o  = tid + rep * 256;                // 1024 outputs
    int i  = o >> 5, j2 = o & 31;
    float v = Rlds[i * 65 + (j2 - i + 31)];
    int a = mode ? i : s;
    int c = mode ? s : i;
    outb[((size_t)(bh * 32 + a) * 32 + c) * 32 + j2] = v;
  }
}

// ---------------------------------------------------------------------------
// Flash attention with relative bias. Block = (128 queries) x (b,h).
// 8 waves, each owns 16 query rows across all keys (no cross-wave softmax).
// K/V tiles staged via async global->LDS b128 (ASYNCcnt), V pre-transposed.
// ---------------------------------------------------------------------------
__global__ __launch_bounds__(256)
void flash_attn_kernel(const unsigned int* __restrict__ qU,  // [bh][l][64 pairs]
                       const unsigned int* __restrict__ kU,  // [bh][l][64 pairs]
                       const unsigned int* __restrict__ vT,  // [bh][d][512 pairs]
                       const float* __restrict__ relw,
                       const float* __restrict__ relh,
                       unsigned short* __restrict__ oH) {    // [b*l][512] bf16
  __shared__ __align__(16) unsigned int kt[64 * 68];         // [key][d-pair]
  __shared__ __align__(16) unsigned int vt[128 * 36];        // [d][key-pair]
  __shared__ __align__(16) unsigned short pts[8][16 * 72];   // per-wave P stage

  const int tid  = threadIdx.x;
  const int lane = tid & 31;
  const int wave = tid >> 5;
  const int half = (lane >> 4) & 1;
  const int ln   = lane & 15;
  const int q0   = blockIdx.x * 128;
  const int bh   = blockIdx.y;
  const int b    = bh >> 2;
  const int h    = bh & 3;

  const unsigned int* qb = qU + (size_t)bh * LSEQ * 64;
  const unsigned int* kb = kU + (size_t)bh * LSEQ * 64;
  const unsigned int* vb = vT + (size_t)bh * KD * 512;

  // Q fragments (A layout) from global, kept in registers for the block.
  v16bf aq[4];
  {
    const unsigned int* qrow = qb + (size_t)(q0 + wave * 16 + ln) * 64;
#pragma unroll
    for (int kk = 0; kk < 4; ++kk) aq[kk] = load_a_frag_row(qrow, kk, lane);
  }

  const v8f zero = {0.f, 0.f, 0.f, 0.f, 0.f, 0.f, 0.f, 0.f};
  v8f oacc[8];
#pragma unroll
  for (int n = 0; n < 8; ++n) oacc[n] = zero;
  float rowmax[8], rowsum[8], rw0[8], rw1[8];
  int baseIdx[8];
#pragma unroll
  for (int r = 0; r < 8; ++r) {
    rowmax[r] = -1e30f;
    rowsum[r] = 0.f;
    int l = q0 + wave * 16 + half * 8 + r;
    int qy = l >> 5, qx = l & 31;
    baseIdx[r] = ((bh * 32 + qy) * 32 + qx) * 32;
    rw0[r] = relw[baseIdx[r] + ln];          // jx = ln     (p even)
    rw1[r] = relw[baseIdx[r] + 16 + ln];     // jx = 16+ln  (p odd)
  }

  for (int j0 = 0; j0 < LSEQ; j0 += 64) {
    __syncthreads();
    if (j0 + 64 < LSEQ)   // pull next K tile toward L2 (global_prefetch_b8)
      __builtin_prefetch(kb + (size_t)(j0 + 64 + (tid >> 2)) * 64, 0, 0);
    // K tile: B-operand layout over d is k's natural [key][d] -> async b128.
#pragma unroll
    for (int i = 0; i < 4; ++i) {
      int q = tid + 256 * i;                 // 64 keys x 16 uint4
      int j = q >> 4, pc4 = (q & 15) * 4;
      async_g2l_b128(kt + j * 68 + pc4, kb + (size_t)(j0 + j) * 64 + pc4);
    }
    // V tile: already transposed globally -> async b128.
#pragma unroll
    for (int i = 0; i < 4; ++i) {
      int q = tid + 256 * i;                 // 128 d x 8 uint4
      int d = q >> 3, c4 = (q & 7) * 4;
      async_g2l_b128(vt + d * 36 + c4,
                     vb + (size_t)d * 512 + (j0 >> 1) + c4);
    }
    wait_async0();
    __syncthreads();

    // S = Q * K^T  (16x64 per wave)
    v8f sacc[4];
#pragma unroll
    for (int p = 0; p < 4; ++p) sacc[p] = zero;
#pragma unroll
    for (int kk = 0; kk < 4; ++kk)
#pragma unroll
      for (int p = 0; p < 4; ++p) {
        v16bf bf = load_b_frag(kt, p * 16, 68, kk * 16, lane);
        sacc[p] = wmma_bf16(aq[kk], bf, sacc[p]);
      }

    // Relative bias: key j = j0+p*16+ln -> jy = jy0 (p<2) or jy0+1 (p>=2).
    const int jy0 = j0 >> 5;
#pragma unroll
    for (int r = 0; r < 8; ++r) {
      float rh0 = relh[baseIdx[r] + jy0];
      float rh1 = relh[baseIdx[r] + jy0 + 1];
      sacc[0][r] += rw0[r] + rh0;
      sacc[1][r] += rw1[r] + rh0;
      sacc[2][r] += rw0[r] + rh1;
      sacc[3][r] += rw1[r] + rh1;
    }

    // Online softmax; row r lives in one 16-lane half (xor 1..8 stays inside).
#pragma unroll
    for (int r = 0; r < 8; ++r) {
      float tm = fmaxf(fmaxf(sacc[0][r], sacc[1][r]),
                       fmaxf(sacc[2][r], sacc[3][r]));
#pragma unroll
      for (int off = 1; off < 16; off <<= 1)
        tm = fmaxf(tm, __shfl_xor(tm, off, 32));
      float nm = fmaxf(rowmax[r], tm);
      float alpha = __expf(rowmax[r] - nm);
      rowmax[r] = nm;
      float ts = 0.f;
#pragma unroll
      for (int p = 0; p < 4; ++p) {
        float pv = __expf(sacc[p][r] - nm);
        sacc[p][r] = pv;
        ts += pv;
      }
#pragma unroll
      for (int off = 1; off < 16; off <<= 1) ts += __shfl_xor(ts, off, 32);
      rowsum[r] = rowsum[r] * alpha + ts;
#pragma unroll
      for (int n = 0; n < 8; ++n) oacc[n][r] *= alpha;
    }

    // Restage P: C/D layout -> A-operand layout through per-wave LDS.
    unsigned short* pp = pts[wave];
#pragma unroll
    for (int p = 0; p < 4; ++p)
#pragma unroll
      for (int r = 0; r < 8; ++r)
        pp[(half * 8 + r) * 72 + p * 16 + ln] = f2bf(sacc[p][r]);
    __syncthreads();

    // O += P * V   (16 queries x 128 d, contraction over 64 keys)
    {
      const unsigned int* prow = (const unsigned int*)(pts[wave] + ln * 72);
#pragma unroll
      for (int kk = 0; kk < 2; ++kk) {
        v16bf ap = load_a_frag_row(prow, kk, lane);  // same pair pattern
#pragma unroll
        for (int n = 0; n < 8; ++n) {
          v16bf bf = load_b_frag(vt, n * 16, 36, kk * 16, lane);
          oacc[n] = wmma_bf16(ap, bf, oacc[n]);
        }
      }
    }
  }

  // Normalize and store bf16 in [b*l][h*128+d] layout for the Wo GEMM.
#pragma unroll
  for (int r = 0; r < 8; ++r) {
    float inv = 1.0f / rowsum[r];
    int l = q0 + wave * 16 + half * 8 + r;
    size_t rowoff = ((size_t)b * LSEQ + l) * 512 + h * 128;
#pragma unroll
    for (int n = 0; n < 8; ++n)
      oH[rowoff + n * 16 + ln] = f2bf(oacc[n][r] * inv);
  }
}

// ---------------------------------------------------------------------------
extern "C" void kernel_launch(void* const* d_in, const int* in_sizes, int n_in,
                              void* d_out, int out_size, void* d_ws,
                              size_t ws_size, hipStream_t stream) {
  const float* x  = (const float*)d_in[0];
  const float* Wq = (const float*)d_in[1];
  const float* Wk = (const float*)d_in[2];
  const float* Wv = (const float*)d_in[3];
  const float* Wo = (const float*)d_in[4];
  const float* pw = (const float*)d_in[5];
  const float* ph = (const float*)d_in[6];

  char* ws = (char*)d_ws;
  size_t off = 0;
  auto alloc = [&](size_t bytes) {
    void* p = ws + off;
    off += (bytes + 255) & ~(size_t)255;
    return p;
  };
  unsigned int* xp  = (unsigned int*)alloc((size_t)8192 * 256 * 4);  // x packed
  unsigned int* Wqt = (unsigned int*)alloc((size_t)512 * 256 * 4);
  unsigned int* Wkt = (unsigned int*)alloc((size_t)512 * 256 * 4);
  unsigned int* Wvt = (unsigned int*)alloc((size_t)512 * 256 * 4);
  unsigned int* Wot = (unsigned int*)alloc((size_t)512 * 256 * 4);
  unsigned short* qbf = (unsigned short*)alloc((size_t)32 * LSEQ * KD * 2);
  unsigned short* kbf = (unsigned short*)alloc((size_t)32 * LSEQ * KD * 2);
  unsigned int* vTp = (unsigned int*)alloc((size_t)32 * KD * 512 * 4);
  unsigned short* obf = (unsigned short*)alloc((size_t)8192 * 512 * 2);
  float* relw = (float*)alloc((size_t)32 * 32 * 32 * 32 * 4);
  float* relh = (float*)alloc((size_t)32 * 32 * 32 * 32 * 4);

  const float scale = 0.08838834764831845f;  // 1/sqrt(128), folded into q

  dim3 blk(256);
  // one-time conversions
  pack_rows_kernel<<<dim3(4096), blk, 0, stream>>>(x, (uint2*)xp);
  dim3 gt(8, 8);
  pack_transpose_kernel<<<gt, blk, 0, stream>>>(Wq, Wqt);
  pack_transpose_kernel<<<gt, blk, 0, stream>>>(Wk, Wkt);
  pack_transpose_kernel<<<gt, blk, 0, stream>>>(Wv, Wvt);
  pack_transpose_kernel<<<gt, blk, 0, stream>>>(Wo, Wot);

  dim3 g1(64, 4);
  gemm_wmma_kernel<0><<<g1, blk, 0, stream>>>(xp, Wqt, scale, qbf, nullptr,
                                              nullptr);
  gemm_wmma_kernel<0><<<g1, blk, 0, stream>>>(xp, Wkt, 1.0f, kbf, nullptr,
                                              nullptr);
  gemm_wmma_kernel<1><<<g1, blk, 0, stream>>>(xp, Wvt, 1.0f, nullptr, vTp,
                                              nullptr);

  rel_logits_kernel<<<dim3(32 * 32, 2), blk, 0, stream>>>(
      (const unsigned int*)qbf, pw, ph, relw, relh);

  flash_attn_kernel<<<dim3(8, 32), blk, 0, stream>>>(
      (const unsigned int*)qbf, (const unsigned int*)kbf, vTp, relw, relh, obf);

  gemm_wmma_kernel<2><<<g1, blk, 0, stream>>>((const unsigned int*)obf, Wot,
                                              1.0f, nullptr, nullptr,
                                              (float*)d_out);
}
